// Listener_83107617177855
// MI455X (gfx1250) — compile-verified, matrix operand order checked by reading, not run on hardware
//
#include <hip/hip_runtime.h>

// Problem dims (match reference)
#define BATCH   512
#define TSTEPS  128
#define HIDDEN  1024
#define VOCAB   1024
#define EMB     512
#define XH      (EMB + HIDDEN)   // 1536
#define OUTN    1024
#define DIMG    2048

typedef __attribute__((ext_vector_type(16))) __bf16 v16bf;
typedef __attribute__((ext_vector_type(8)))  float  v8f;

union FragBF { v16bf v; uint4 u[2]; };
union AccF   { v8f v; float f[8]; };

// ---------- helpers ----------
__device__ __forceinline__ unsigned short f2bf(float f) {
    unsigned u = __float_as_uint(f);
    u += 0x7FFFu + ((u >> 16) & 1u);       // round-to-nearest-even
    return (unsigned short)(u >> 16);
}
__device__ __forceinline__ float sigmf(float x) {
    return 1.0f / (1.0f + __expf(-x));
}
__device__ __forceinline__ float tanh_fast(float x) {
    return 1.0f - 2.0f / (__expf(2.0f * x) + 1.0f);
}

__device__ __forceinline__ v8f wmma_bf16(v16bf a, v16bf b, v8f c) {
    return __builtin_amdgcn_wmma_f32_16x16x32_bf16(
        false, a, false, b, (short)0, c, false, false);
}

// Fragment fetch from a precomputed per-lane base pointer.
// Base pointer already encodes (row0 + lane%16)*ld + (lane/16)*8; the k0 index
// becomes an instruction-immediate offset after unrolling.
__device__ __forceinline__ v16bf frag_at(const unsigned short* __restrict__ p, int k0) {
    FragBF f;
    f.u[0] = *reinterpret_cast<const uint4*>(p + k0);        // K = k0..k0+7
    f.u[1] = *reinterpret_cast<const uint4*>(p + k0 + 16);   // K = k0+16..k0+23
    return f.v;
}

// ---------- prep kernels ----------
__global__ void k_transpose_bf16(const float* __restrict__ src,
                                 unsigned short* __restrict__ dst, int K, int N) {
    long i = (long)blockIdx.x * blockDim.x + threadIdx.x;
    long total = (long)K * N;
    if (i >= total) return;
    int n = (int)(i / K);
    int k = (int)(i % K);
    dst[i] = f2bf(src[(size_t)k * N + n]);
}

__global__ void k_f32_to_bf16(const float* __restrict__ src,
                              unsigned short* __restrict__ dst, int n) {
    int i = blockIdx.x * blockDim.x + threadIdx.x;
    if (i < n) dst[i] = f2bf(src[i]);
}

__global__ void k_init_state(float* __restrict__ c, unsigned short* __restrict__ xh0) {
    int i = blockIdx.x * blockDim.x + threadIdx.x;
    if (i >= BATCH * HIDDEN) return;
    c[i] = 0.0f;
    int b = i / HIDDEN, n = i % HIDDEN;
    xh0[(size_t)b * XH + EMB + n] = 0;
}

__global__ void k_gather_x(const float* __restrict__ embed,
                           const int* __restrict__ message, int t,
                           unsigned short* __restrict__ xh) {
    int b = blockIdx.x;
    int idx = message[(size_t)b * TSTEPS + t];
    const float* src = embed + (size_t)idx * EMB;
    unsigned short* dst = xh + (size_t)b * XH;
    for (int j = threadIdx.x; j < EMB; j += blockDim.x)
        dst[j] = f2bf(src[j]);
}

// ---------- fused LSTM step ----------
// Wave tile: 32M x 16N x 4 gates (8 WMMAs / 6 fragment loads per K-step).
// Block: 256 thr = 8 waves as 2(M) x 4(N) -> 64x64 tile. Grid (16, 8).
// Pipelined via cur/nxt fragment sets; all base addresses hoisted to pointers.
struct LFrags { v16bf a0, a1, wi, wg, wf, wo; };

__global__ void __launch_bounds__(256)
k_lstm_step(const unsigned short* __restrict__ xh,     // [512,1536] bf16
            const unsigned short* __restrict__ Wt,     // [4096,1536] bf16 (N-major)
            const float* __restrict__ b_cell,          // [4096]
            float* __restrict__ c,                     // [512,1024]
            unsigned short* __restrict__ xh_next,      // h -> [:,512:1536]
            unsigned short* __restrict__ hbf) {        // [512,1024]
    int wave = threadIdx.x >> 5;
    int lane = threadIdx.x & 31;
    int m0 = blockIdx.y * 64 + (wave >> 2) * 32;
    int n0 = blockIdx.x * 64 + (wave & 3) * 16;

    // per-lane base pointers (hoisted out of the K loop)
    int lrow = lane & 15;
    int koff = (lane >> 4) << 3;
    const unsigned short* pa0 = xh + (size_t)(m0 + lrow) * XH + koff;
    const unsigned short* pa1 = pa0 + (size_t)16 * XH;
    const unsigned short* pwi = Wt + (size_t)(n0 + lrow) * XH + koff;
    const unsigned short* pwg = pwi + (size_t)HIDDEN * XH;
    const unsigned short* pwf = pwi + (size_t)2 * HIDDEN * XH;
    const unsigned short* pwo = pwi + (size_t)3 * HIDDEN * XH;

    v8f ai0 = {}, ai1 = {}, ag0 = {}, ag1 = {};
    v8f af0 = {}, af1 = {}, ao0 = {}, ao1 = {};

    LFrags cur;
    cur.a0 = frag_at(pa0, 0); cur.a1 = frag_at(pa1, 0);
    cur.wi = frag_at(pwi, 0); cur.wg = frag_at(pwg, 0);
    cur.wf = frag_at(pwf, 0); cur.wo = frag_at(pwo, 0);

#pragma unroll 2
    for (int k0 = 32; k0 < XH; k0 += 32) {
        LFrags nxt;
        nxt.a0 = frag_at(pa0, k0); nxt.a1 = frag_at(pa1, k0);
        nxt.wi = frag_at(pwi, k0); nxt.wg = frag_at(pwg, k0);
        nxt.wf = frag_at(pwf, k0); nxt.wo = frag_at(pwo, k0);

        ai0 = wmma_bf16(cur.a0, cur.wi, ai0);
        ai1 = wmma_bf16(cur.a1, cur.wi, ai1);
        ag0 = wmma_bf16(cur.a0, cur.wg, ag0);
        ag1 = wmma_bf16(cur.a1, cur.wg, ag1);
        af0 = wmma_bf16(cur.a0, cur.wf, af0);
        af1 = wmma_bf16(cur.a1, cur.wf, af1);
        ao0 = wmma_bf16(cur.a0, cur.wo, ao0);
        ao1 = wmma_bf16(cur.a1, cur.wo, ao1);
        cur = nxt;
    }
    ai0 = wmma_bf16(cur.a0, cur.wi, ai0);
    ai1 = wmma_bf16(cur.a1, cur.wi, ai1);
    ag0 = wmma_bf16(cur.a0, cur.wg, ag0);
    ag1 = wmma_bf16(cur.a1, cur.wg, ag1);
    af0 = wmma_bf16(cur.a0, cur.wf, af0);
    af1 = wmma_bf16(cur.a1, cur.wf, af1);
    ao0 = wmma_bf16(cur.a0, cur.wo, ao0);
    ao1 = wmma_bf16(cur.a1, cur.wo, ao1);

    int col = n0 + lrow;
    float Bi = b_cell[col];
    float Bg = b_cell[col + HIDDEN];
    float Bf = b_cell[col + 2 * HIDDEN];
    float Bo = b_cell[col + 3 * HIDDEN];

    AccF ui[2], ug[2], uf[2], uo[2];
    ui[0].v = ai0; ui[1].v = ai1;
    ug[0].v = ag0; ug[1].v = ag1;
    uf[0].v = af0; uf[1].v = af1;
    uo[0].v = ao0; uo[1].v = ao1;

#pragma unroll
    for (int mt = 0; mt < 2; ++mt) {
        int rbase = m0 + mt * 16 + ((lane >> 4) << 3);
#pragma unroll
        for (int r = 0; r < 8; ++r) {
            int row = rbase + r;
            size_t cix = (size_t)row * HIDDEN + col;
            float iv = sigmf(ui[mt].f[r] + Bi);
            float gv = tanh_fast(ug[mt].f[r] + Bg);
            float fv = sigmf(uf[mt].f[r] + Bf + 1.0f);
            float ov = sigmf(uo[mt].f[r] + Bo);
            float cn = fv * c[cix] + iv * gv;
            float hn = ov * tanh_fast(cn);
            c[cix] = cn;
            unsigned short hb = f2bf(hn);
            xh_next[(size_t)row * XH + EMB + col] = hb;
            hbf[cix] = hb;
        }
    }
}

// ---------- generic bf16 GEMM + bias -> fp32 out ----------
// Wave tile 32x32 (4 WMMAs / 4 fragment loads per K-step), pipelined.
// Block: 8 waves as 2(M) x 4(N) -> 64x128 tile. Grid (N/128, M/64).
struct GFrags { v16bf a0, a1, b0, b1; };

__global__ void __launch_bounds__(256)
k_gemm_bias(const unsigned short* __restrict__ A,   // [M,K] bf16
            const unsigned short* __restrict__ Bt,  // [N,K] bf16 (N-major)
            const float* __restrict__ bias,         // [N]
            float* __restrict__ out,                // [M,N]
            int N, int K) {
    int wave = threadIdx.x >> 5;
    int lane = threadIdx.x & 31;
    int m0 = blockIdx.y * 64  + (wave >> 2) * 32;
    int n0 = blockIdx.x * 128 + (wave & 3) * 32;

    int lrow = lane & 15;
    int koff = (lane >> 4) << 3;
    const unsigned short* pa0 = A  + (size_t)(m0 + lrow) * K + koff;
    const unsigned short* pa1 = pa0 + (size_t)16 * K;
    const unsigned short* pb0 = Bt + (size_t)(n0 + lrow) * K + koff;
    const unsigned short* pb1 = pb0 + (size_t)16 * K;

    v8f c00 = {}, c01 = {}, c10 = {}, c11 = {};

    GFrags cur;
    cur.a0 = frag_at(pa0, 0); cur.a1 = frag_at(pa1, 0);
    cur.b0 = frag_at(pb0, 0); cur.b1 = frag_at(pb1, 0);

#pragma unroll 2
    for (int k0 = 32; k0 < K; k0 += 32) {
        GFrags nxt;
        nxt.a0 = frag_at(pa0, k0); nxt.a1 = frag_at(pa1, k0);
        nxt.b0 = frag_at(pb0, k0); nxt.b1 = frag_at(pb1, k0);

        c00 = wmma_bf16(cur.a0, cur.b0, c00);
        c01 = wmma_bf16(cur.a0, cur.b1, c01);
        c10 = wmma_bf16(cur.a1, cur.b0, c10);
        c11 = wmma_bf16(cur.a1, cur.b1, c11);
        cur = nxt;
    }
    c00 = wmma_bf16(cur.a0, cur.b0, c00);
    c01 = wmma_bf16(cur.a0, cur.b1, c01);
    c10 = wmma_bf16(cur.a1, cur.b0, c10);
    c11 = wmma_bf16(cur.a1, cur.b1, c11);

    AccF u[2][2];
    u[0][0].v = c00; u[0][1].v = c01; u[1][0].v = c10; u[1][1].v = c11;

#pragma unroll
    for (int nt = 0; nt < 2; ++nt) {
        int col = n0 + nt * 16 + lrow;
        float bv = bias[col];
#pragma unroll
        for (int mt = 0; mt < 2; ++mt) {
            int rbase = m0 + mt * 16 + ((lane >> 4) << 3);
#pragma unroll
            for (int r = 0; r < 8; ++r)
                out[(size_t)(rbase + r) * N + col] = u[mt][nt].f[r] + bv;
        }
    }
}

// ---------- host ----------
extern "C" void kernel_launch(void* const* d_in, const int* in_sizes, int n_in,
                              void* d_out, int out_size, void* d_ws, size_t ws_size,
                              hipStream_t stream) {
    const float* images = (const float*)d_in[0];
    const float* embed  = (const float*)d_in[1];
    const float* W_cell = (const float*)d_in[2];
    const float* b_cell = (const float*)d_in[3];
    const float* W_img  = (const float*)d_in[4];
    const float* b_img  = (const float*)d_in[5];
    const float* W_hid  = (const float*)d_in[6];
    const float* b_hid  = (const float*)d_in[7];
    const int*   message = (const int*)d_in[8];

    char* p = (char*)d_ws;
    unsigned short* Wc_t   = (unsigned short*)p; p += (size_t)4 * HIDDEN * XH * 2;
    unsigned short* Wi_t   = (unsigned short*)p; p += (size_t)OUTN * DIMG * 2;
    unsigned short* Wh_t   = (unsigned short*)p; p += (size_t)OUTN * HIDDEN * 2;
    unsigned short* img_bf = (unsigned short*)p; p += (size_t)BATCH * DIMG * 2;
    unsigned short* xhA    = (unsigned short*)p; p += (size_t)BATCH * XH * 2;
    unsigned short* xhB    = (unsigned short*)p; p += (size_t)BATCH * XH * 2;
    unsigned short* hbf    = (unsigned short*)p; p += (size_t)BATCH * HIDDEN * 2;
    float*          cbuf   = (float*)p;

    // 1) weights -> N-major bf16 (L2-resident across all 128 steps)
    {
        long n;
        n = (long)XH * 4 * HIDDEN;
        k_transpose_bf16<<<(n + 255) / 256, 256, 0, stream>>>(W_cell, Wc_t, XH, 4 * HIDDEN);
        n = (long)DIMG * OUTN;
        k_transpose_bf16<<<(n + 255) / 256, 256, 0, stream>>>(W_img, Wi_t, DIMG, OUTN);
        n = (long)HIDDEN * OUTN;
        k_transpose_bf16<<<(n + 255) / 256, 256, 0, stream>>>(W_hid, Wh_t, HIDDEN, OUTN);
        n = (long)BATCH * DIMG;
        k_f32_to_bf16<<<(n + 255) / 256, 256, 0, stream>>>(images, img_bf, (int)n);
    }

    // 2) image encoder (independent of LSTM)
    k_gemm_bias<<<dim3(OUTN / 128, BATCH / 64), 256, 0, stream>>>(
        img_bf, Wi_t, b_img, (float*)d_out, OUTN, DIMG);

    // 3) LSTM: init state, gather x_0, 128 pipelined fused steps
    k_init_state<<<(BATCH * HIDDEN + 255) / 256, 256, 0, stream>>>(cbuf, xhA);
    k_gather_x<<<BATCH, 256, 0, stream>>>(embed, message, 0, xhA);

    unsigned short* cur = xhA;
    unsigned short* nxt = xhB;
    for (int t = 0; t < TSTEPS; ++t) {
        k_lstm_step<<<dim3(HIDDEN / 64, BATCH / 64), 256, 0, stream>>>(
            cur, Wc_t, b_cell, cbuf, nxt, hbf);
        if (t + 1 < TSTEPS)
            k_gather_x<<<BATCH, 256, 0, stream>>>(embed, message, t + 1, nxt);
        unsigned short* tmp = cur; cur = nxt; nxt = tmp;
    }

    // 4) hidden encoder from final h
    k_gemm_bias<<<dim3(OUTN / 128, BATCH / 64), 256, 0, stream>>>(
        hbf, Wh_t, b_hid, (float*)d_out + (size_t)BATCH * OUTN, OUTN, HIDDEN);
}
